// GaussianUpsampling_45294725103828
// MI455X (gfx1250) — matrix operand order repaired and checked by reading, not compiled, not run on hardware
//
#include <hip/hip_runtime.h>

typedef __attribute__((ext_vector_type(16))) _Float16 v16h;
typedef __attribute__((ext_vector_type(8)))  _Float16 v8h;
typedef __attribute__((ext_vector_type(8)))  float    v8f;
typedef int v4i __attribute__((__vector_size__(16)));   // matches builtin param type

#define AS1 __attribute__((address_space(1)))
#define AS3 __attribute__((address_space(3)))

#define B_SZ   2
#define TTEXT  1024
#define ADIM   512
#define TFEATS 16384
#define DELTA  0.1f
#define KC     64                 // K chunk = 2 k-steps of 16x16x32 f16 WMMA
#define MWG    128                // frames per workgroup (8 M-tiles of 16)
#define NCHUNK (TTEXT / KC)       // 16

#if __has_builtin(__builtin_amdgcn_global_load_async_to_lds_b128) && \
    __has_builtin(__builtin_amdgcn_s_wait_asynccnt)
#define USE_ASYNC 1
#else
#define USE_ASYNC 0
#endif

// ---------------------------------------------------------------------------
// Kernel 1: token centers c[b,j] = cumsum(d)[j] - d[j]/2  (Hillis-Steele scan)
// ---------------------------------------------------------------------------
__global__ void gu_centers_kernel(const int* __restrict__ ds, float* __restrict__ cent) {
    __shared__ float s[TTEXT];
    const int b = blockIdx.x;
    const int j = threadIdx.x;
    const float d = (float)ds[b * TTEXT + j];
    s[j] = d;
    __syncthreads();
    #pragma unroll
    for (int off = 1; off < TTEXT; off <<= 1) {
        float v = (j >= off) ? s[j - off] : 0.0f;
        __syncthreads();
        s[j] += v;
        __syncthreads();
    }
    cent[b * TTEXT + j] = s[j] - 0.5f * d;
}

// ---------------------------------------------------------------------------
// Kernel 2: one-shot convert + transpose: hT[b][d][k] = (f16) hs[b][k][d]
// ---------------------------------------------------------------------------
__global__ __launch_bounds__(256)
void gu_cvt_kernel(const float* __restrict__ hs, _Float16* __restrict__ hT) {
    __shared__ _Float16 tile[32][33];  // +1 pad: no bank conflicts
    const int b  = blockIdx.z;
    const int k0 = blockIdx.y * 32;
    const int d0 = blockIdx.x * 32;
    const int tx = threadIdx.x;        // 0..31
    const int ty = threadIdx.y;        // 0..7
    #pragma unroll
    for (int j = 0; j < 32; j += 8)
        tile[ty + j][tx] = (_Float16)hs[((size_t)b * TTEXT + k0 + ty + j) * ADIM + d0 + tx];
    __syncthreads();
    #pragma unroll
    for (int j = 0; j < 32; j += 8)
        hT[((size_t)b * ADIM + d0 + ty + j) * TTEXT + k0 + tx] = tile[tx][ty + j];
}

// ---------------------------------------------------------------------------
// Kernel 3: fused softmax(energy) @ hs via f16 WMMA, f32 accumulation.
// 32 waves arranged 8(M) x 4(N); wave owns 16 frames x 128 cols = 8 acc tiles.
// Double-buffered LDS, async global->LDS B staging, one barrier per chunk.
// ---------------------------------------------------------------------------
__global__ __launch_bounds__(1024)
void gu_main_kernel(const _Float16* __restrict__ hT, const float* __restrict__ cent,
                    float* __restrict__ out) {
    __shared__ float    c_lds[TTEXT];           // 4 KB
    __shared__ _Float16 a_lds[2][MWG][KC];      // 2 x 16 KB weight tiles (A)
    __shared__ _Float16 h_lds[2][ADIM][KC];     // 2 x 64 KB B chunks, N-major
    __shared__ float    sinv_lds[MWG];          // 0.5 KB

    const int b    = blockIdx.y;
    const int t0   = blockIdx.x * MWG;
    const int tid  = threadIdx.x;
    const int lane = tid & 31;
    const int wave = tid >> 5;
    const int mrow = wave >> 2;                 // 0..7: 16-frame row
    const int ngrp = wave & 3;                  // 0..3: 128-col group

    for (int i = tid; i < TTEXT; i += 1024) c_lds[i] = cent[b * TTEXT + i];
    __syncthreads();

    // A-operand lane map (16-bit A 16x32); B-operand lane map (16-bit B 32x16)
    const int arow  = mrow * 16 + (lane & 15);
    const int khalf = (lane >> 4) * 8;
    const int kbase = (lane >> 4) * 16;
    const int colb  = ngrp * 128 + (lane & 15);

    // energy map: thread owns (frame fl, k-segment kseg) for the whole K-loop
    const int   fl   = tid >> 3;                // 0..127
    const int   kseg = (tid & 7) * 8;           // 0..56
    const float tf   = (float)(t0 + fl);

    // B staging map: thread copies 64 B (32 halfs) of one dim-row per chunk
    const int bd = tid >> 1;                    // 0..511
    const int bo = (tid & 1) * 32;              // half offset 0 / 32

    v8f   acc[8] = {};
    float sreg   = 0.0f;

    auto stageA = [&](int klo, int buf) {
        v8h wv;
        #pragma unroll
        for (int j = 0; j < 8; j++) {
            const float c  = c_lds[klo + kseg + j];
            const float dd = tf - c;
            const float w  = __expf(-DELTA * dd * dd);
            wv[j] = (_Float16)w;
            sreg += w;
        }
        *(v8h*)&a_lds[buf][fl][kseg] = wv;      // one ds_store_b128
    };

    auto stageB = [&](int klo, int buf) {
        const _Float16* src = hT + ((size_t)b * ADIM + bd) * TTEXT + klo + bo;
#if USE_ASYNC
        AS1 v4i* g0 = (AS1 v4i*)src;
        AS3 v4i* l0 = (AS3 v4i*)&h_lds[buf][bd][bo];
        __builtin_amdgcn_global_load_async_to_lds_b128(g0, l0, 0, 0);
        __builtin_amdgcn_global_load_async_to_lds_b128(g0, l0, 16, 0);
        __builtin_amdgcn_global_load_async_to_lds_b128(g0, l0, 32, 0);
        __builtin_amdgcn_global_load_async_to_lds_b128(g0, l0, 48, 0);
#else
        if (klo + KC < TTEXT)
            __builtin_prefetch(src + KC, 0, 1);
        *(uint4*)&h_lds[buf][bd][bo]      = *(const uint4*)(src);
        *(uint4*)&h_lds[buf][bd][bo + 8]  = *(const uint4*)(src + 8);
        *(uint4*)&h_lds[buf][bd][bo + 16] = *(const uint4*)(src + 16);
        *(uint4*)&h_lds[buf][bd][bo + 24] = *(const uint4*)(src + 24);
#endif
    };

    // prologue: stage chunk 0 into buffer 0
    stageA(0, 0);
    stageB(0, 0);

    for (int i = 0; i < NCHUNK; i++) {
        const int p = i & 1;
#if USE_ASYNC
        __builtin_amdgcn_s_wait_asynccnt(0);    // own B(i) copies complete
#endif
        __syncthreads();                        // all waves' A(i)/B(i) visible;
                                                // prior LDS reads retired (WAR-safe)
        if (i + 1 < NCHUNK) {                   // overlap next-chunk staging
            stageA((i + 1) * KC, 1 - p);
            stageB((i + 1) * KC, 1 - p);
        }

        // ---- 16 WMMAs on buffer p: 2 k-steps x 8 N-tiles ----
        union V16 { v16h v; v8h h[2]; };
        #pragma unroll
        for (int ks = 0; ks < 2; ks++) {
            const int kk = ks * 32;
            V16 Au;
            Au.h[0] = *(const v8h*)&a_lds[p][arow][kk + khalf];
            Au.h[1] = *(const v8h*)&a_lds[p][arow][kk + 16 + khalf];
            #pragma unroll
            for (int nt = 0; nt < 8; nt++) {
                V16 Bu;
                Bu.h[0] = *(const v8h*)&h_lds[p][colb + nt * 16][kk + khalf];
                Bu.h[1] = *(const v8h*)&h_lds[p][colb + nt * 16][kk + 16 + khalf];
                acc[nt] = __builtin_amdgcn_wmma_f32_16x16x32_f16(
                    false, Au.v, false, Bu.v, (short)0, acc[nt], false, false);
            }
        }
    }
    __syncthreads();

    // ---- reduce softmax denominators (reuse a_lds as f32 scratch) ----
    float* part = (float*)&a_lds[0][0][0];      // 128 x 8 f32 = 4 KB
    part[fl * 8 + (tid & 7)] = sreg;
    __syncthreads();
    if (tid < MWG) {
        float s = 0.0f;
        #pragma unroll
        for (int j = 0; j < 8; j++) s += part[tid * 8 + j];
        sinv_lds[tid] = 1.0f / s;
    }
    __syncthreads();

    // ---- epilogue: scale + store ----
    // C/D layout: VGPR r, lanes 0-15 -> M=r, lanes 16-31 -> M=r+8; N = lane&15
    const int mb = (lane >> 4) * 8;
    float sv[8];
    #pragma unroll
    for (int r = 0; r < 8; r++) sv[r] = sinv_lds[mrow * 16 + mb + r];

    #pragma unroll
    for (int nt = 0; nt < 8; nt++) {
        float* op = out + ((size_t)b * TFEATS + t0 + mrow * 16 + mb) * ADIM
                        + ngrp * 128 + nt * 16 + (lane & 15);
        #pragma unroll
        for (int r = 0; r < 8; r++)
            op[(size_t)r * ADIM] = acc[nt][r] * sv[r];
    }
}

// ---------------------------------------------------------------------------
extern "C" void kernel_launch(void* const* d_in, const int* in_sizes, int n_in,
                              void* d_out, int out_size, void* d_ws, size_t ws_size,
                              hipStream_t stream) {
    const float* hs  = (const float*)d_in[0];     // (2,1024,512) f32
    const int*   ds  = (const int*)d_in[1];       // (2,1024) i32
    float*       out = (float*)d_out;             // (2,16384,512) f32

    float*    cent = (float*)d_ws;                        // 8 KB
    _Float16* hT   = (_Float16*)((char*)d_ws + 8192);     // 2 MB f16 transposed

    gu_centers_kernel<<<dim3(B_SZ), dim3(TTEXT), 0, stream>>>(ds, cent);

    dim3 cgrid(ADIM / 32, TTEXT / 32, B_SZ);
    gu_cvt_kernel<<<cgrid, dim3(32, 8), 0, stream>>>(hs, hT);

    dim3 grid(TFEATS / MWG, B_SZ);
    gu_main_kernel<<<grid, dim3(1024), 0, stream>>>(hT, cent, out);
}
